// SKConv2d_31808527794809
// MI455X (gfx1250) — matrix operand order repaired and checked by reading, not compiled, not run on hardware
//
#include <hip/hip_runtime.h>

// ---------------------------------------------------------------------------
// SKConv2d collapsed to a single bf16-WMMA im2col GEMM for gfx1250 (MI455X).
//   out = im2col(x) @ Weff + bias,  Weff = (sum_t S1 U1 + U2^T S2) / (2T)
// ---------------------------------------------------------------------------

typedef __attribute__((ext_vector_type(16))) __bf16 v16bf;
typedef __attribute__((ext_vector_type(8)))  __bf16 bf16x8;
typedef __attribute__((ext_vector_type(8)))  float  v8f;

#define B_      16
#define CIN     256
#define H_      56
#define W_      56
#define COUT    256
#define T_      6
#define R_      8
#define KR      72
#define KCIN    2304          // 256*9
#define HP      58
#define WP      58
#define HW      (H_*W_)       // 3136
#define NTOT    (B_*HW)       // 50176  (= 392 * 128 exactly)
#define OUT_BSTRIDE (COUT*HW) // 802816

#define WFRAG_ELEMS (KCIN*COUT)       // 589824 bf16
#define WFRAG_BYTES (WFRAG_ELEMS*2)   // 1179648 (512B aligned)
#define XPAD_ELEMS  (B_*HP*WP*CIN)    // 13778944
#define XPAD_BYTES  (XPAD_ELEMS*2)    // 27557888

// ---------------------------------------------------------------------------
// Pre-pass 1: x [B,CIN,H,W] f32  ->  xpad [B,HP,WP,CIN] bf16 (pad=1, NHWC).
// Reads coalesced along w; border is zeroed by hipMemsetAsync beforehand.
// ---------------------------------------------------------------------------
__global__ void pad_nhwc_bf16(const float* __restrict__ x,
                              __bf16* __restrict__ xpad) {
    int tid = blockIdx.x * blockDim.x + threadIdx.x;   // (b,c,h,w), w fastest
    int w  = tid % W_;
    int t1 = tid / W_;
    int h  = t1 % H_;
    int t2 = t1 / H_;
    int c  = t2 % CIN;
    int b  = t2 / CIN;
    float v = x[tid];
    xpad[(((b*HP + (h+1))*WP + (w+1))*CIN) + c] = (__bf16)v;
}

// ---------------------------------------------------------------------------
// Pre-pass 2: build effective weight, K reordered to (kh,kw,c), scaled by
// 1/(2T), stored bf16 in a WMMA B-fragment layout with contiguous k (kin = 16*half + e, as A uses):
//   idx = ((kblk*16 + oblk)*32 + lane)*16 + e
//   lane = (kin/16)*16 + (o & 15),  e = kin % 16,  kin = k' % 32
// so each lane's 16 B elements are 32 contiguous bytes.
// ---------------------------------------------------------------------------
__global__ void build_wfrag(const float* __restrict__ S1s,
                            const float* __restrict__ S2s,
                            const float* __restrict__ U1s,
                            const float* __restrict__ U2s,
                            __bf16* __restrict__ wfrag) {
    int tid = blockIdx.x * blockDim.x + threadIdx.x;   // k'*256 + o
    int o  = tid & 255;
    int kp = tid >> 8;          // k' = p*256 + c   (p = kh*3+kw)
    int p  = kp >> 8;
    int c  = kp & 255;
    int k  = c*9 + p;           // original (c,kh,kw) flat index

    float acc = 0.0f;
    for (int t = 0; t < T_; ++t) {
        const float* s1 = S1s + (t*KCIN + k)*R_;
        const float* u1 = U1s + t*R_*COUT + o;
#pragma unroll
        for (int r = 0; r < R_; ++r) acc += s1[r] * u1[r*COUT];
        const float* u2 = U2s + t*KR*KCIN + k;
        const float* s2 = S2s + t*KR*COUT + o;
        for (int q = 0; q < KR; ++q) acc += u2[q*KCIN] * s2[q*COUT];
    }
    acc *= (1.0f / (2.0f * T_));

    int kblk = kp >> 5;
    int kin  = kp & 31;
    int oblk = o >> 4;
    int lane = ((kin >> 4) << 4) | (o & 15);
    int e    = kin & 15;
    wfrag[((((kblk*16 + oblk)*32) + lane) << 4) + e] = (__bf16)acc;
}

// ---------------------------------------------------------------------------
// Main GEMM: C[50176, 256] = A[50176, 2304] @ Weff[2304, 256]
// Block: 256 thr = 8 waves (2M x 4N), tile 128M x 256N. Wave tile 64x64:
// 4 A frags, 4 B frags, 16 accumulators, 72 K-steps of 32.
// ---------------------------------------------------------------------------
__global__ void __launch_bounds__(256)
skconv_gemm(const __bf16* __restrict__ xpad,
            const __bf16* __restrict__ wfrag,
            const float*  __restrict__ bias,
            float* __restrict__ out) {
    const int lane = threadIdx.x & 31;
    const int wave = threadIdx.x >> 5;
    const int wm   = wave >> 2;        // 0..1
    const int wn   = wave & 3;         // 0..3
    const int half = lane >> 4;        // K-half within fragment
    const int l16  = lane & 15;

    const int waveM = blockIdx.x * 128 + wm * 64;

    // Per-lane A base pointers: lane row m -> (b,h,w); base points at the
    // patch's top-left in padded coords (pad folded in).
    const __bf16* abase[4];
#pragma unroll
    for (int mi = 0; mi < 4; ++mi) {
        int m  = waveM + mi*16 + l16;
        int b  = m / HW;
        int hw = m - b*HW;
        int h  = hw / W_;
        int w  = hw - h*W_;
        abase[mi] = xpad + (((b*HP + h)*WP + w) * CIN);
    }
    // Per-lane B base: + kk*8192 + ni*512 each step (32B contiguous / lane).
    const __bf16* bbase = wfrag + (wn*4)*512 + lane*16;

    v8f acc[4][4];
#pragma unroll
    for (int mi = 0; mi < 4; ++mi)
#pragma unroll
        for (int ni = 0; ni < 4; ++ni)
            acc[mi][ni] = (v8f)(0.0f);

    for (int kk = 0; kk < 72; ++kk) {       // K-slice = (kh,kw) tap p, c-block cb
        int p  = kk >> 3;
        int cb = kk & 7;
        int kh = p / 3;
        int kw = p - kh*3;
        int aoff = (kh*WP + kw)*CIN + cb*32 + half*16;

        v16bf a[4];
#pragma unroll
        for (int mi = 0; mi < 4; ++mi) {
            bf16x8 lo = *(const bf16x8*)(abase[mi] + aoff);       // K = klo..klo+7
            bf16x8 hi = *(const bf16x8*)(abase[mi] + aoff + 8);   // K = klo+8..+15
#pragma unroll
            for (int i = 0; i < 8; ++i) { a[mi][i] = lo[i]; a[mi][i+8] = hi[i]; }
        }

        v16bf bf[4];
        const __bf16* bp = bbase + kk*8192;
#pragma unroll
        for (int ni = 0; ni < 4; ++ni) {
            bf16x8 lo = *(const bf16x8*)(bp + ni*512);
            bf16x8 hi = *(const bf16x8*)(bp + ni*512 + 8);
#pragma unroll
            for (int i = 0; i < 8; ++i) { bf[ni][i] = lo[i]; bf[ni][i+8] = hi[i]; }
        }

#pragma unroll
        for (int mi = 0; mi < 4; ++mi)
#pragma unroll
            for (int ni = 0; ni < 4; ++ni)
                acc[mi][ni] = __builtin_amdgcn_wmma_f32_16x16x32_bf16(
                    false, a[mi], false, bf[ni], (short)0, acc[mi][ni],
                    false, false);
    }

    // Epilogue: +bias, scatter back to NCHW. D layout: lane col = l16,
    // rows = half*8 + v.
    int moff[4][8];
#pragma unroll
    for (int mi = 0; mi < 4; ++mi)
#pragma unroll
        for (int v = 0; v < 8; ++v) {
            int m  = waveM + mi*16 + half*8 + v;
            int b  = m / HW;
            int hw = m - b*HW;
            int h  = hw / W_;
            int w  = hw - h*W_;
            moff[mi][v] = b*OUT_BSTRIDE + h*W_ + w;
        }

#pragma unroll
    for (int ni = 0; ni < 4; ++ni) {
        int o = wn*64 + ni*16 + l16;
        float bo = bias[o];
        int obase = o*HW;
#pragma unroll
        for (int mi = 0; mi < 4; ++mi)
#pragma unroll
            for (int v = 0; v < 8; ++v)
                __builtin_nontemporal_store(acc[mi][ni][v] + bo,
                                            &out[moff[mi][v] + obase]);
    }
}

// ---------------------------------------------------------------------------
extern "C" void kernel_launch(void* const* d_in, const int* in_sizes, int n_in,
                              void* d_out, int out_size, void* d_ws, size_t ws_size,
                              hipStream_t stream) {
    const float* x    = (const float*)d_in[0];
    const float* S1s  = (const float*)d_in[1];
    const float* S2s  = (const float*)d_in[2];
    const float* U1s  = (const float*)d_in[3];
    const float* U2s  = (const float*)d_in[4];
    const float* bias = (const float*)d_in[5];
    float* out = (float*)d_out;

    __bf16* wfrag = (__bf16*)d_ws;                            // 1.18 MB
    __bf16* xpad  = (__bf16*)((char*)d_ws + WFRAG_BYTES);     // 27.6 MB

    // zero padded image border (interior overwritten); capturable stream op
    hipMemsetAsync(xpad, 0, XPAD_BYTES, stream);

    build_wfrag<<<(KCIN*COUT)/256, 256, 0, stream>>>(S1s, S2s, U1s, U2s, wfrag);
    pad_nhwc_bf16<<<(B_*CIN*H_*W_)/256, 256, 0, stream>>>(x, xpad);
    skconv_gemm<<<NTOT/128, 256, 0, stream>>>(xpad, wfrag, bias, out);
}
